// DynamicGRU_5970004542181
// MI455X (gfx1250) — matrix operand order repaired and checked by reading, not compiled, not run on hardware
//
#include <hip/hip_runtime.h>

// ---------------------------------------------------------------------------
// AUGRU (DIEN) fused persistent kernel for gfx1250 (MI455X).
// B=1024, T=200, D=128, U=128, K = D+U = 256.
//
// One block = 16 batch rows, 8 wave32s, each wave owns a 16-wide N-slice of U.
// Weights (bf16, transposed) + h (f32 + bf16 copy) live in LDS for all 200
// steps. Per step, per wave: 24 v_wmma_f32_16x16x32_bf16.
//
// Latency structure (the recurrence is the critical path):
//   - x tile + att are double-buffered in LDS and software-pipelined through
//     registers: global NT loads for t+1 issue at the top of iteration t,
//     the LDS stores happen at the bottom -> 2 barriers/step, loads fully
//     hidden under compute.
//   - candidate x-part hoisted before the r*h barrier; post-barrier matrix
//     work is 2 independent WMMA chains of depth 2.
// ---------------------------------------------------------------------------

typedef __attribute__((ext_vector_type(16))) __bf16 v16bf;
typedef __attribute__((ext_vector_type(8)))  __bf16 v8bf;
typedef __attribute__((ext_vector_type(8)))  float  v8f;

constexpr int kB = 1024;
constexpr int kT = 200;
constexpr int kD = 128;
constexpr int kU = 128;
constexpr int kK = kD + kU;              // 256

constexpr int WT_STRIDE   = kK + 8;      // 264 halves: rows 16B-aligned, bank-rotated
constexpr int TILE_STRIDE = kD + 8;      // 136 halves
constexpr int XBUF        = 16 * TILE_STRIDE;

constexpr size_t kSmemBytes =
    (size_t)(3 * kU * WT_STRIDE        // bf16 W^T x3
             + 2 * XBUF                // x tile, double buffered
             + 2 * XBUF) * 2           // h_bf + rh_bf
    + (size_t)16 * kU * 4              // h_f32
    + 2 * 16 * 4                       // att, double buffered
    + 16 * 4;                          // seq_len

// A-fragment (16x32 bf16, MxK), row stride TILE_STRIDE.
// Lanes 0-15 (g=0): halves 0-7 = K+0..7, halves 8-15 = K+16..23;
// lanes 16-31 (g=1): K+8..15 and K+24..31.
__device__ __forceinline__ v16bf frag_a(const __bf16* base, int k0, int lane) {
  int m = lane & 15, g = lane >> 4;
  const __bf16* p = base + m * TILE_STRIDE + k0 + 8 * g;
  v8bf lo = *(const v8bf*)p;
  v8bf hi = *(const v8bf*)(p + 16);
  v16bf r;
#pragma unroll
  for (int i = 0; i < 8; ++i) { r[i] = lo[i]; r[i + 8] = hi[i]; }
  return r;
}

// B-fragment (32x16 bf16, KxN) from W^T stored [n][k], row stride WT_STRIDE.
// Lane group g holds K+8g..8g+7 and K+16+8g..16+8g+7; column n = n0+(lane&15).
__device__ __forceinline__ v16bf frag_b(const __bf16* wt, int n0, int k0, int lane) {
  const __bf16* p = wt + (n0 + (lane & 15)) * WT_STRIDE + k0 + 8 * (lane >> 4);
  v8bf lo = *(const v8bf*)p;
  v8bf hi = *(const v8bf*)(p + 16);
  v16bf r;
#pragma unroll
  for (int i = 0; i < 8; ++i) { r[i] = lo[i]; r[i + 8] = hi[i]; }
  return r;
}

__device__ __forceinline__ float sigmoidf_(float x) {
  return 1.0f / (1.0f + __expf(-x));
}

#define WMMA_BF16(A, B, C) \
  __builtin_amdgcn_wmma_f32_16x16x32_bf16(false, (A), false, (B), (short)0, (C), false, false)

__global__ void augru_fused_kernel(const float* __restrict__ seq_emb,
                                   const long long* __restrict__ seq_len,
                                   const float* __restrict__ att,
                                   const float* __restrict__ Wu,
                                   const float* __restrict__ Wr,
                                   const float* __restrict__ Wc,
                                   float* __restrict__ out) {
  extern __shared__ __align__(16) char smem_raw[];
  __bf16* wtu   = (__bf16*)smem_raw;                  // [128][264]
  __bf16* wtr   = wtu + kU * WT_STRIDE;
  __bf16* wtc   = wtr + kU * WT_STRIDE;
  __bf16* x_bf  = wtc + kU * WT_STRIDE;               // [2][16][136]
  __bf16* h_bf  = x_bf + 2 * XBUF;                    // [16][136]
  __bf16* rh_bf = h_bf + XBUF;                        // [16][136]
  float*  h_f32 = (float*)(rh_bf + XBUF);             // [16][128]
  float*  att_l = h_f32 + 16 * kU;                    // [2][16]
  int*    sl_l  = (int*)(att_l + 32);                 // [16]

  const int tid  = threadIdx.x;
  const int lane = tid & 31;
  const int wv   = tid >> 5;       // 8 waves
  const int n0   = wv * 16;        // this wave's N-slice of U
  const int g    = lane >> 4;
  const int nc   = lane & 15;
  const int b0   = blockIdx.x * 16;

  // ---- one-time setup: transpose+convert weights to bf16 in LDS ----
  const float* wsrc[3] = {Wu, Wr, Wc};
  __bf16*      wdst[3] = {wtu, wtr, wtc};
#pragma unroll
  for (int gi = 0; gi < 3; ++gi) {
    const float* src = wsrc[gi];
    __bf16* dst = wdst[gi];
    for (int i = tid; i < kK * kU; i += 256) {        // coalesced global reads
      int k = i >> 7, n = i & 127;                    // W is [256][128] row-major
      dst[n * WT_STRIDE + k] = (__bf16)src[i];
    }
  }
  for (int i = tid; i < XBUF; i += 256) {
    h_bf[i]  = (__bf16)0.0f;
    rh_bf[i] = (__bf16)0.0f;
  }
  for (int i = tid; i < 16 * kU; i += 256) h_f32[i] = 0.0f;
  if (tid < 16) sl_l[tid] = (int)seq_len[b0 + tid];   // seq_length is int64 [B,1]

  // ---- prologue: stage x/att for t=0 into buffer 0 ----
  float xpre[8];
  float apre = 0.0f;
#pragma unroll
  for (int q = 0; q < 8; ++q) {
    int i = tid + q * 256;
    int m = i >> 7, k = i & 127;
    xpre[q] = __builtin_nontemporal_load(&seq_emb[((size_t)(b0 + m) * kT) * kD + k]);
  }
  if (tid < 16) apre = att[(size_t)(b0 + tid) * kT];
#pragma unroll
  for (int q = 0; q < 8; ++q) {
    int i = tid + q * 256;
    int m = i >> 7, k = i & 127;
    x_bf[m * TILE_STRIDE + k] = (__bf16)xpre[q];
  }
  if (tid < 16) att_l[tid] = apre;

  // ---- recurrence over t ----
  for (int t = 0; t < kT; ++t) {
    const int cur = t & 1;
    const int nxt = cur ^ 1;
    __syncthreads();  // buf[cur] staged; h updates visible; buf[nxt] free

    // issue global prefetch for t+1 (held in registers until iteration end)
    const int tp = (t + 1 < kT) ? (t + 1) : t;
#pragma unroll
    for (int q = 0; q < 8; ++q) {
      int i = tid + q * 256;
      int m = i >> 7, k = i & 127;
      xpre[q] = __builtin_nontemporal_load(
          &seq_emb[((size_t)(b0 + m) * kT + tp) * kD + k]);
    }
    if (tid < 16) apre = att[(size_t)(b0 + tid) * kT + tp];

    const __bf16* xb = x_bf + cur * XBUF;

    // phase c: u, r over full K; candidate x-part (2 chains). 20 WMMAs.
    v8f accu = {}, accr = {}, acc0 = {}, acc1 = {};
#pragma unroll
    for (int k0 = 0; k0 < 64; k0 += 32) {
      v16bf a = frag_a(xb, k0, lane);
      accu = WMMA_BF16(a, frag_b(wtu, n0, k0, lane), accu);
      accr = WMMA_BF16(a, frag_b(wtr, n0, k0, lane), accr);
      acc0 = WMMA_BF16(a, frag_b(wtc, n0, k0, lane), acc0);
    }
#pragma unroll
    for (int k0 = 64; k0 < kD; k0 += 32) {
      v16bf a = frag_a(xb, k0, lane);
      accu = WMMA_BF16(a, frag_b(wtu, n0, k0, lane), accu);
      accr = WMMA_BF16(a, frag_b(wtr, n0, k0, lane), accr);
      acc1 = WMMA_BF16(a, frag_b(wtc, n0, k0, lane), acc1);
    }
#pragma unroll
    for (int k0 = 0; k0 < kU; k0 += 32) {
      v16bf a = frag_a(h_bf, k0, lane);
      accu = WMMA_BF16(a, frag_b(wtu, n0, kD + k0, lane), accu);
      accr = WMMA_BF16(a, frag_b(wtr, n0, kD + k0, lane), accr);
    }

    float ug[8];
#pragma unroll
    for (int j = 0; j < 8; ++j) {
      ug[j]    = sigmoidf_(accu[j]);
      float rg = sigmoidf_(accr[j]);
      int m = j + 8 * g;
      rh_bf[m * TILE_STRIDE + n0 + nc] =
          (__bf16)(rg * h_f32[m * kU + n0 + nc]);
    }
    __syncthreads();  // rh complete; att_l[cur] visible

    // candidate rh-part: 2 independent chains of depth 2, then combine
#pragma unroll
    for (int k0 = 0; k0 < 64; k0 += 32) {
      acc0 = WMMA_BF16(frag_a(rh_bf, k0, lane),
                       frag_b(wtc, n0, kD + k0, lane), acc0);
      acc1 = WMMA_BF16(frag_a(rh_bf, 64 + k0, lane),
                       frag_b(wtc, n0, kD + 64 + k0, lane), acc1);
    }
    v8f accc = acc0 + acc1;

    // h update + masked freeze + output
#pragma unroll
    for (int j = 0; j < 8; ++j) {
      int m = j + 8 * g;
      int n = n0 + nc;
      float ht   = tanhf(accc[j]);
      float ut   = ug[j] * att_l[cur * 16 + m];
      float hold = h_f32[m * kU + n];
      float hnew = (1.0f - ut) * hold + ut * ht;
      float hfin = (t < sl_l[m]) ? hnew : hold;
      h_f32[m * kU + n] = hfin;
      h_bf[m * TILE_STRIDE + n] = (__bf16)hfin;
      __builtin_nontemporal_store(
          hfin, &out[((size_t)(b0 + m) * kT + t) * kU + n]);
    }

    // stage prefetched t+1 data into the nxt buffers (loads now long done)
    __bf16* xn = x_bf + nxt * XBUF;
#pragma unroll
    for (int q = 0; q < 8; ++q) {
      int i = tid + q * 256;
      int m = i >> 7, k = i & 127;
      xn[m * TILE_STRIDE + k] = (__bf16)xpre[q];
    }
    if (tid < 16) att_l[nxt * 16 + tid] = apre;
  }
}

extern "C" void kernel_launch(void* const* d_in, const int* in_sizes, int n_in,
                              void* d_out, int out_size, void* d_ws, size_t ws_size,
                              hipStream_t stream) {
  const float*     seq_emb = (const float*)d_in[0];
  const long long* seq_len = (const long long*)d_in[1];   // int64 [B,1]
  const float*     att     = (const float*)d_in[2];
  const float*     Wu      = (const float*)d_in[3];
  const float*     Wr      = (const float*)d_in[4];
  const float*     Wc      = (const float*)d_in[5];
  float*           out     = (float*)d_out;

  (void)in_sizes; (void)n_in; (void)out_size; (void)d_ws; (void)ws_size;

  hipFuncSetAttribute(reinterpret_cast<const void*>(augru_fused_kernel),
                      hipFuncAttributeMaxDynamicSharedMemorySize,
                      (int)kSmemBytes);

  augru_fused_kernel<<<dim3(kB / 16), dim3(256), kSmemBytes, stream>>>(
      seq_emb, seq_len, att, Wu, Wr, Wc, out);
}
